// MixtureOfExperts_89902255440747
// MI455X (gfx1250) — matrix operand order repaired and checked by reading, not compiled, not run on hardware
//
#include <hip/hip_runtime.h>
#include <hip/hip_bf16.h>

#define T_TOK 4096
#define H_DIM 1024
#define E_NUM 8
#define F_DIM 4096
#define TOPK  2

typedef __attribute__((ext_vector_type(16))) __bf16 v16bf;
typedef __attribute__((ext_vector_type(8)))  float  v8f;

union V16 { v16bf v; uint4 q[2]; };

// ---------------------------------------------------------------------------
// Fragment loaders (wave32, 16x16x32 bf16 WMMA layouts per CDNA5 ISA 7.12.2)
// A (16x32, MxK): lane r=lane&15 holds row r; hb=lane>>4 selects
//   K chunks [hb*8, hb*8+8) and [16+hb*8, ...). Two 16B loads.
// B (32x16, KxN): lane n=lane&15 holds column n (= a row of W for x@W^T);
//   hb selects K chunks [hb*8, hb*8+8) and [16+hb*8, ...). Two 16B loads.
// ---------------------------------------------------------------------------
__device__ __forceinline__ v16bf load_frag_a(const __bf16* rowBase, int k, int hb) {
  V16 r;
  r.q[0] = *(const uint4*)(rowBase + k + hb * 8);
  r.q[1] = *(const uint4*)(rowBase + k + 16 + hb * 8);
  return r.v;
}

__device__ __forceinline__ v16bf load_frag_b(const __bf16* rowBase, int k, int hb) {
  V16 r;
  const __bf16* p = rowBase + k + hb * 8;
  r.q[0] = *(const uint4*)(p);
  r.q[1] = *(const uint4*)(p + 16);
  return r.v;
}

__device__ __forceinline__ v8f wmma_bf16(v16bf a, v16bf b, v8f c) {
  return __builtin_amdgcn_wmma_f32_16x16x32_bf16(false, a, false, b, (short)0, c,
                                                 false, false);
}

// ---------------------------------------------------------------------------
// Utility kernels
// ---------------------------------------------------------------------------
__global__ void moe_zero_f32(float* __restrict__ p, int n) {
  int i = blockIdx.x * blockDim.x + threadIdx.x;
  if (i < n) p[i] = 0.0f;
}

__global__ void moe_zero_i32(int* __restrict__ p, int n) {
  int i = blockIdx.x * blockDim.x + threadIdx.x;
  if (i < n) p[i] = 0;
}

// f32 -> bf16 (packed stores), used for x and the three weight tensors.
__global__ void moe_cvt_bf16(const float* __restrict__ src, __bf16* __restrict__ dst,
                             int n4) {
  int i = blockIdx.x * blockDim.x + threadIdx.x;
  if (i < n4) {
    float4 v = ((const float4*)src)[i];
    union { __bf16 b[4]; uint2 u; } o;
    o.b[0] = (__bf16)v.x; o.b[1] = (__bf16)v.y;
    o.b[2] = (__bf16)v.z; o.b[3] = (__bf16)v.w;
    ((uint2*)dst)[i] = o.u;
  }
}

// ---------------------------------------------------------------------------
// Gating: logits -> softmax -> top-2 -> renormalized gates, compacted
// per-expert token lists via atomic slot allocation.
// ---------------------------------------------------------------------------
__global__ __launch_bounds__(256)
void moe_gate(const float* __restrict__ x, const float* __restrict__ gate_w,
              int* __restrict__ counts, int* __restrict__ tok,
              float* __restrict__ gw) {
  __shared__ float gsh[E_NUM * H_DIM];  // 32 KB
  for (int i = threadIdx.x; i < E_NUM * H_DIM; i += 256) gsh[i] = gate_w[i];
  __syncthreads();

  const int t = blockIdx.x * 256 + threadIdx.x;
  const float* xr = x + (size_t)t * H_DIM;

  float acc[E_NUM];
#pragma unroll
  for (int e = 0; e < E_NUM; e++) acc[e] = 0.0f;

  for (int h = 0; h < H_DIM; h += 4) {
    float4 xv = *(const float4*)(xr + h);
#pragma unroll
    for (int e = 0; e < E_NUM; e++) {
      const float* g = gsh + e * H_DIM + h;
      acc[e] += xv.x * g[0] + xv.y * g[1] + xv.z * g[2] + xv.w * g[3];
    }
  }

  // softmax over E=8 (TEMPERATURE == 1)
  float m = acc[0];
#pragma unroll
  for (int e = 1; e < E_NUM; e++) m = fmaxf(m, acc[e]);
  float p[E_NUM], s = 0.0f;
#pragma unroll
  for (int e = 0; e < E_NUM; e++) { p[e] = __expf(acc[e] - m); s += p[e]; }

  // top-2 (ties -> lowest index, matching jax.lax.top_k)
  int i1 = 0;
#pragma unroll
  for (int e = 1; e < E_NUM; e++) if (p[e] > p[i1]) i1 = e;
  int i2 = (i1 == 0) ? 1 : 0;
#pragma unroll
  for (int e = 0; e < E_NUM; e++) if (e != i1 && p[e] > p[i2]) i2 = e;

  float g1 = p[i1] / s, g2 = p[i2] / s;
  float denom = g1 + g2 + 1e-8f;
  g1 /= denom; g2 /= denom;

  int s1 = atomicAdd(&counts[i1], 1);
  tok[i1 * T_TOK + s1] = t;
  gw [i1 * T_TOK + s1] = g1;
  int s2 = atomicAdd(&counts[i2], 1);
  tok[i2 * T_TOK + s2] = t;
  gw [i2 * T_TOK + s2] = g2;
}

// Prefix offsets + load-balance loss -> d_out[T*H]
__global__ void moe_finalize(const int* __restrict__ counts, int* __restrict__ offs,
                             float* __restrict__ loss_out) {
  if (threadIdx.x == 0) {
    int o = 0;
    for (int e = 0; e < E_NUM; e++) { offs[e] = o; o += counts[e]; }
    offs[E_NUM] = o;
    float total = (float)o;
    float mean = 1.0f / (float)E_NUM;  // usage sums to 1
    float var = 0.0f;
    for (int e = 0; e < E_NUM; e++) {
      float u = (float)counts[e] / total;
      float d = u - mean;
      var += d * d;
    }
    var /= (float)(E_NUM - 1);  // ddof = 1
    float lb = var / (mean + 1e-8f);
    loss_out[0] = lb * lb;
  }
}

// ---------------------------------------------------------------------------
// Up projection + SwiGLU:  act[slot,f] = gate * g * silu(u),  bf16 out.
// Per-wave register tile: 32 tokens x 32 F, both g and u -> 8 WMMAs per
// K=32 step on 2 A-frags + 4 B-frags (all bf16 b128 loads, no in-loop cvt).
// Block: 8 waves = 32 tokens x 256 F.
// ---------------------------------------------------------------------------
__global__ __launch_bounds__(256)
void moe_up_swiglu(const __bf16* __restrict__ Wgb, const __bf16* __restrict__ Wub,
                   const int* __restrict__ counts, const int* __restrict__ offs,
                   const int* __restrict__ tok, const float* __restrict__ gw,
                   const __bf16* __restrict__ xb, __bf16* __restrict__ act) {
  const int e = blockIdx.z;
  const int cnt = counts[e];
  const int trow0 = blockIdx.y * 32;
  if (trow0 >= cnt) return;

  const int wave = threadIdx.x >> 5;
  const int lane = threadIdx.x & 31;
  const int hb = lane >> 4;
  const int r  = lane & 15;
  const int f0 = blockIdx.x * 256 + wave * 32;  // this wave's 32-wide F strip

  // Two gathered A rows per lane: trow0+r and trow0+16+r (padded rows compute
  // on a valid token, never stored).
  const int base = e * T_TOK;
  const int ar0 = trow0 + r;
  const int ar1 = trow0 + 16 + r;
  const int tk0 = (ar0 < cnt) ? tok[base + ar0] : tok[base + trow0];
  const int tk1 = (ar1 < cnt) ? tok[base + ar1] : tok[base + trow0];
  const __bf16* a0p = xb + (size_t)tk0 * H_DIM;
  const __bf16* a1p = xb + (size_t)tk1 * H_DIM;

  const __bf16* g0p = Wgb + ((size_t)e * F_DIM + f0 + r) * H_DIM;
  const __bf16* g1p = g0p + (size_t)16 * H_DIM;
  const __bf16* u0p = Wub + ((size_t)e * F_DIM + f0 + r) * H_DIM;
  const __bf16* u1p = u0p + (size_t)16 * H_DIM;

  v8f cg[2][2] = {{{}, {}}, {{}, {}}};
  v8f cu[2][2] = {{{}, {}}, {{}, {}}};

  for (int k = 0; k < H_DIM; k += 32) {
    v16bf a0  = load_frag_a(a0p, k, hb);
    v16bf a1  = load_frag_a(a1p, k, hb);
    v16bf bg0 = load_frag_b(g0p, k, hb);
    v16bf bg1 = load_frag_b(g1p, k, hb);
    v16bf bu0 = load_frag_b(u0p, k, hb);
    v16bf bu1 = load_frag_b(u1p, k, hb);
    cg[0][0] = wmma_bf16(a0, bg0, cg[0][0]);
    cg[0][1] = wmma_bf16(a0, bg1, cg[0][1]);
    cg[1][0] = wmma_bf16(a1, bg0, cg[1][0]);
    cg[1][1] = wmma_bf16(a1, bg1, cg[1][1]);
    cu[0][0] = wmma_bf16(a0, bu0, cu[0][0]);
    cu[0][1] = wmma_bf16(a0, bu1, cu[0][1]);
    cu[1][0] = wmma_bf16(a1, bu0, cu[1][0]);
    cu[1][1] = wmma_bf16(a1, bu1, cu[1][1]);
  }

  // C/D layout: element j of lane -> row m = j + 8*hb (+16 for mi=1),
  // col n = r (+16 for ni=1).
  const size_t rowBase = (size_t)(offs[e] + trow0);
#pragma unroll
  for (int mi = 0; mi < 2; mi++) {
#pragma unroll
    for (int j = 0; j < 8; j++) {
      const int m = mi * 16 + j + 8 * hb;
      if (trow0 + m < cnt) {
        const float gate = gw[base + trow0 + m];
        const size_t rowOff = (rowBase + m) * F_DIM + f0 + r;
#pragma unroll
        for (int ni = 0; ni < 2; ni++) {
          float g = cg[mi][ni][j];
          float u = cu[mi][ni][j];
          float sig = __builtin_amdgcn_rcpf(1.0f + __expf(-u));  // sigmoid(u)
          act[rowOff + ni * 16] = (__bf16)(g * u * sig * gate);
        }
      }
    }
  }
}

// ---------------------------------------------------------------------------
// Down projection + scatter accumulate: out[t,:] += act[slot,:] @ Wd[e]^T
// Per-wave register tile: 32 tokens x 32 H -> 4 WMMAs per K=32 step on
// 2 A-frags + 2 B-frags (all bf16). Block: 8 waves = 32 tokens x 256 H.
// ---------------------------------------------------------------------------
__global__ __launch_bounds__(256)
void moe_down(const __bf16* __restrict__ Wdb,
              const int* __restrict__ counts, const int* __restrict__ offs,
              const int* __restrict__ tok,
              const __bf16* __restrict__ act, float* __restrict__ out) {
  const int e = blockIdx.z;
  const int cnt = counts[e];
  const int trow0 = blockIdx.y * 32;
  if (trow0 >= cnt) return;

  const int wave = threadIdx.x >> 5;
  const int lane = threadIdx.x & 31;
  const int hb = lane >> 4;
  const int r  = lane & 15;
  const int h0 = blockIdx.x * 256 + wave * 32;

  const int base = e * T_TOK;
  const int ar0 = trow0 + r;
  const int ar1 = trow0 + 16 + r;
  const int sr0 = offs[e] + ((ar0 < cnt) ? ar0 : trow0);
  const int sr1 = offs[e] + ((ar1 < cnt) ? ar1 : trow0);
  const __bf16* a0p = act + (size_t)sr0 * F_DIM;
  const __bf16* a1p = act + (size_t)sr1 * F_DIM;

  const __bf16* b0p = Wdb + ((size_t)e * H_DIM + h0 + r) * F_DIM;
  const __bf16* b1p = b0p + (size_t)16 * F_DIM;

  v8f c[2][2] = {{{}, {}}, {{}, {}}};

  for (int k = 0; k < F_DIM; k += 32) {
    v16bf a0 = load_frag_a(a0p, k, hb);
    v16bf a1 = load_frag_a(a1p, k, hb);
    v16bf b0 = load_frag_b(b0p, k, hb);
    v16bf b1 = load_frag_b(b1p, k, hb);
    c[0][0] = wmma_bf16(a0, b0, c[0][0]);
    c[0][1] = wmma_bf16(a0, b1, c[0][1]);
    c[1][0] = wmma_bf16(a1, b0, c[1][0]);
    c[1][1] = wmma_bf16(a1, b1, c[1][1]);
  }

#pragma unroll
  for (int mi = 0; mi < 2; mi++) {
#pragma unroll
    for (int j = 0; j < 8; j++) {
      const int m = mi * 16 + j + 8 * hb;
      if (trow0 + m < cnt) {
        const int t = tok[base + trow0 + m];
        float* orow = out + (size_t)t * H_DIM + h0 + r;
#pragma unroll
        for (int ni = 0; ni < 2; ni++) {
          __hip_atomic_fetch_add(orow + ni * 16, c[mi][ni][j],
                                 __ATOMIC_RELAXED, __HIP_MEMORY_SCOPE_AGENT);
        }
      }
    }
  }
}

// ---------------------------------------------------------------------------
extern "C" void kernel_launch(void* const* d_in, const int* in_sizes, int n_in,
                              void* d_out, int out_size, void* d_ws, size_t ws_size,
                              hipStream_t stream) {
  const float* x      = (const float*)d_in[0];
  const float* gate_w = (const float*)d_in[1];
  const float* Wg     = (const float*)d_in[2];
  const float* Wu     = (const float*)d_in[3];
  const float* Wd     = (const float*)d_in[4];
  float* out = (float*)d_out;

  // Workspace carve-up (~265 MB total)
  char* ws = (char*)d_ws;
  size_t off = 0;
  auto alloc = [&](size_t bytes) -> void* {
    off = (off + 255) & ~(size_t)255;
    void* p = ws + off;
    off += bytes;
    return p;
  };
  const size_t W_ELEMS = (size_t)E_NUM * F_DIM * H_DIM;  // 33.5M per tensor
  int*    counts = (int*)   alloc(E_NUM * sizeof(int));
  int*    offs   = (int*)   alloc((E_NUM + 1) * sizeof(int));
  int*    tok    = (int*)   alloc((size_t)E_NUM * T_TOK * sizeof(int));      // 128 KB
  float*  gw     = (float*) alloc((size_t)E_NUM * T_TOK * sizeof(float));    // 128 KB
  __bf16* xb     = (__bf16*)alloc((size_t)T_TOK * H_DIM * sizeof(__bf16));   // 8 MB
  __bf16* act    = (__bf16*)alloc((size_t)T_TOK * TOPK * F_DIM * sizeof(__bf16)); // 64 MB
  __bf16* Wgb    = (__bf16*)alloc(W_ELEMS * sizeof(__bf16));                 // 64 MB
  __bf16* Wub    = (__bf16*)alloc(W_ELEMS * sizeof(__bf16));                 // 64 MB
  __bf16* Wdb    = (__bf16*)alloc(W_ELEMS * sizeof(__bf16));                 // 64 MB
  (void)ws_size; (void)in_sizes; (void)n_in; (void)out_size;

  // 1) init: zero accumulator output and counts; convert x + weights -> bf16
  moe_zero_i32<<<1, 32, 0, stream>>>(counts, E_NUM);
  moe_zero_f32<<<(T_TOK * H_DIM) / 256, 256, 0, stream>>>(out, T_TOK * H_DIM);
  moe_cvt_bf16<<<(T_TOK * H_DIM / 4) / 256, 256, 0, stream>>>(x, xb, T_TOK * H_DIM / 4);
  const int w4 = (int)(W_ELEMS / 4);
  moe_cvt_bf16<<<w4 / 256, 256, 0, stream>>>(Wg, Wgb, w4);
  moe_cvt_bf16<<<w4 / 256, 256, 0, stream>>>(Wu, Wub, w4);
  moe_cvt_bf16<<<w4 / 256, 256, 0, stream>>>(Wd, Wdb, w4);

  // 2) gating + routing
  moe_gate<<<T_TOK / 256, 256, 0, stream>>>(x, gate_w, counts, tok, gw);

  // 3) offsets + load-balance loss (last output element)
  moe_finalize<<<1, 32, 0, stream>>>(counts, offs, out + (size_t)T_TOK * H_DIM);

  // 4) up projection + SwiGLU (gathered tokens, 32x32 register tiles, WMMA bf16)
  dim3 gUp(F_DIM / 256, T_TOK / 32, E_NUM);
  moe_up_swiglu<<<gUp, 256, 0, stream>>>(Wgb, Wub, counts, offs, tok, gw, xb, act);

  // 5) down projection + scatter-add (32x32 register tiles, WMMA bf16, f32 atomics)
  dim3 gDn(H_DIM / 256, T_TOK / 32, E_NUM);
  moe_down<<<gDn, 256, 0, stream>>>(Wdb, counts, offs, tok, act, out);
}
